// DCNConv_27333171872448
// MI455X (gfx1250) — matrix operand order repaired and checked by reading, not compiled, not run on hardware
//
#include <hip/hip_runtime.h>
#include <hip/hip_bf16.h>

typedef __attribute__((ext_vector_type(16))) __bf16         v16bf;
typedef __attribute__((ext_vector_type(8)))  float          v8f;
typedef __attribute__((ext_vector_type(4)))  float          v4f;
typedef __attribute__((ext_vector_type(8)))  unsigned int   v8u;
typedef __attribute__((ext_vector_type(4)))  unsigned int   v4u;
typedef __attribute__((ext_vector_type(4)))  unsigned short v4us;

#define N_   8
#define C_   128
#define H_   64
#define W_   64
#define K2_  9
#define P_   256
#define CK   1152      // C_*K2_ : GEMM K dimension (reordered as kk*128+c)
#define L_   4096      // H_*W_ pixels per image (Ho=Wo=64)
#define PIX_TILE 16    // pixels per workgroup (GEMM N tile)
#define PADK  1168     // LDS row stride in ushorts (1152 + 16 pad; 32B-aligned, bank-rotating)
#define KSTEP 32       // WMMA bf16 K per instruction

__device__ __forceinline__ unsigned short f2bf(float f) {
  unsigned int u = __float_as_uint(f);
  u += 0x7FFFu + ((u >> 16) & 1u);   // round-to-nearest-even
  return (unsigned short)(u >> 16);
}

// ---------------------------------------------------------------------------
// Kernel 0a: fp32 weights (P, c*9+kk) -> bf16 weights (P, kk*128+c)
// ---------------------------------------------------------------------------
__global__ __launch_bounds__(256) void wconvert_kernel(const float* __restrict__ w,
                                                       unsigned short* __restrict__ wbf) {
  int gid = blockIdx.x * 256 + threadIdx.x;      // P_*CK total, exact multiple of 256
  int p = gid / CK, k = gid - p * CK;
  int kk = k >> 7, c = k & 127;
  wbf[gid] = f2bf(w[p * CK + c * K2_ + kk]);
}

// ---------------------------------------------------------------------------
// Kernel 0b: NCHW -> NHWC transpose of x (LDS-tiled, both sides coalesced).
// Block: (n, 64-pixel strip). LDS tile 64 x 130 floats.
// ---------------------------------------------------------------------------
__global__ __launch_bounds__(256) void xpose_kernel(const float* __restrict__ x,
                                                    float* __restrict__ xT) {
  __shared__ float t[64 * 130];
  const int n   = blockIdx.y;
  const int hw0 = blockIdx.x * 64;
  const int tid = threadIdx.x;
  {
    int hw = tid & 63, cb = tid >> 6;            // cb 0..3
#pragma unroll 8
    for (int cc = 0; cc < 32; ++cc) {
      int c = (cc << 2) + cb;
      t[hw * 130 + c] = x[(((size_t)n * C_ + c) << 12) + hw0 + hw];  // coalesced in hw
    }
  }
  __syncthreads();
  {
    int c = tid & 127, hq = tid >> 7;            // hq 0..1
#pragma unroll 8
    for (int i = 0; i < 32; ++i) {
      int hw = (i << 1) + hq;
      xT[((size_t)(n * L_ + hw0 + hw) << 7) + c] = t[hw * 130 + c]; // coalesced in c
    }
  }
}

// ---------------------------------------------------------------------------
// Kernel 1: 3x3 conv (pad 1), 18 offset + 9 mask channels.
// One thread per (n, pixel): x taps read ONCE, all 27 output channels
// accumulated in registers; weight reads are wave-uniform -> scalar loads.
// mask output already has 2*sigmoid applied.
// ---------------------------------------------------------------------------
__global__ __launch_bounds__(256) void offmask_conv_kernel(
    const float* __restrict__ x, const float* __restrict__ offset_w,
    const float* __restrict__ offset_b, const float* __restrict__ mask_w,
    const float* __restrict__ mask_b, float* __restrict__ offbuf,
    float* __restrict__ maskbuf) {
  int gid = blockIdx.x * 256 + threadIdx.x;      // N_*L_ threads
  int n = gid >> 12, pix = gid & (L_ - 1);
  int yy = pix >> 6, xx = pix & 63;

  int   off[9];
  float okf[9];
#pragma unroll
  for (int t = 0; t < 9; ++t) {
    int iy = yy - 1 + t / 3, ix = xx - 1 + t % 3;
    bool ok = (iy >= 0 && iy < H_ && ix >= 0 && ix < W_);
    off[t] = ok ? ((iy << 6) + ix) : 0;
    okf[t] = ok ? 1.f : 0.f;
  }

  float acc[27];
#pragma unroll
  for (int o = 0; o < 27; ++o) acc[o] = 0.f;

  const float* xb = x + (((size_t)n * C_) << 12);
  for (int c = 0; c < C_; ++c) {
    const float* xc = xb + ((size_t)c << 12);
    float xv[9];
#pragma unroll
    for (int t = 0; t < 9; ++t) xv[t] = okf[t] * xc[off[t]];   // coalesced in pix
    const float* wo = offset_w + c * 9;                        // wave-uniform
    const float* wm = mask_w + c * 9;                          // wave-uniform
#pragma unroll
    for (int o = 0; o < 18; ++o)
#pragma unroll
      for (int t = 0; t < 9; ++t) acc[o] += wo[(size_t)o * CK + t] * xv[t];
#pragma unroll
    for (int o = 0; o < 9; ++o)
#pragma unroll
      for (int t = 0; t < 9; ++t) acc[18 + o] += wm[(size_t)o * CK + t] * xv[t];
  }

  size_t nb = ((size_t)n * 18) << 12;
#pragma unroll
  for (int o = 0; o < 18; ++o)
    offbuf[nb + ((size_t)o << 12) + pix] = acc[o] + offset_b[o];
  size_t mb = ((size_t)n * 9) << 12;
#pragma unroll
  for (int o = 0; o < 9; ++o)
    maskbuf[mb + ((size_t)o << 12) + pix] =
        2.f / (1.f + __expf(-(acc[18 + o] + mask_b[o])));
}

// ---------------------------------------------------------------------------
// Kernel 2: fused bilinear gather (NHWC -> LDS, bf16) + WMMA GEMM.
// Workgroup: batch n, 16-pixel tile. 256 threads = 8 waves.
// Wave w computes output rows [32w, 32w+32) x 16 pixels via two 16x16 tiles.
// ---------------------------------------------------------------------------
__global__ __launch_bounds__(256) void dcn_fused_kernel(
    const float* __restrict__ xT, const unsigned short* __restrict__ wbf,
    const float* __restrict__ offbuf, const float* __restrict__ maskbuf,
    const float* __restrict__ bias, float* __restrict__ out) {
  __shared__ __align__(32) unsigned short colLDS[PIX_TILE * PADK];  // ~37 KB
  __shared__ float pyL[PIX_TILE * K2_], pxL[PIX_TILE * K2_], mL[PIX_TILE * K2_];

  const int tid  = threadIdx.x;
  const int n    = blockIdx.y;
  const int pix0 = blockIdx.x * PIX_TILE;

  // ---- Phase 1: per (pixel, tap) sample coordinates + mask ----
  if (tid < PIX_TILE * K2_) {
    int pix = tid & 15, kk = tid >> 4;
    int pg = pix0 + pix;
    int yy = pg >> 6, xx = pg & 63;
    float dy = offbuf[((size_t)(n * 18 + 2 * kk) << 12) + pg];
    float dx = offbuf[((size_t)(n * 18 + 2 * kk + 1) << 12) + pg];
    pyL[tid] = dy + (float)(yy - 1 + kk / 3);
    pxL[tid] = dx + (float)(xx - 1 + kk % 3);
    mL[tid]  = maskbuf[((size_t)(n * 9 + kk) << 12) + pg];
  }
  __syncthreads();

  // ---- Phase 2: bilinear gather into LDS. Each wave is uniform in
  // (pixel, tap); 32 lanes span the 128 channels as float4 -> each corner is
  // one contiguous, coalesced 512B read from NHWC x; LDS store is 8B/lane. ----
  const float* xTn = xT + (((size_t)n) << 12) * C_;
#pragma unroll 2
  for (int it = 0; it < (PIX_TILE * K2_ * (C_ / 4)) / 256; ++it) {   // 18 iters
    int idx  = tid + (it << 8);
    int pair = idx >> 5;             // kk*16 + pix : wave-uniform
    int c0   = (idx & 31) << 2;      // channel base, 4 per lane
    float py = pyL[pair], px = pxL[pair], mm = mL[pair];
    float fy = floorf(py), fx = floorf(px);
    int   y0 = (int)fy,    x0 = (int)fx;
    float ly = py - fy,    lx = px - fx;
    v4f v = {0.f, 0.f, 0.f, 0.f};
    auto corner = [&](int yi, int xi, float w) {
      if (yi >= 0 && yi < H_ && xi >= 0 && xi < W_) {
        v4f t = *(const v4f*)(xTn + ((size_t)((yi << 6) + xi) << 7) + c0);
        v += w * t;
      }
    };
    corner(y0,     x0,     (1.f - ly) * (1.f - lx));
    corner(y0,     x0 + 1, (1.f - ly) * lx);
    corner(y0 + 1, x0,     ly * (1.f - lx));
    corner(y0 + 1, x0 + 1, ly * lx);
    int pix = pair & 15, kk = pair >> 4;
    v4us o;
#pragma unroll
    for (int j = 0; j < 4; ++j) o[j] = f2bf(v[j] * mm);
    *(v4us*)&colLDS[pix * PADK + (kk << 7) + c0] = o;
  }
  __syncthreads();

  // ---- Phase 3: WMMA GEMM  D(16x16,f32) += A(16x32,bf16) x B(32x16,bf16) ----
  const int lane = tid & 31;
  const int wv   = tid >> 5;       // 0..7
  const int half = lane >> 4;      // 0/1
  const int row  = lane & 15;      // A row / B col / D col

  v8f acc0, acc1;
#pragma unroll
  for (int i = 0; i < 8; ++i) { acc0[i] = 0.f; acc1[i] = 0.f; }

  // A in this code's contiguous k order (same as B): lanes hold row=lane%16; VGPR0-3 -> K = half*16+j,
  // VGPR4-7 -> K = half*16+8+j  => one contiguous 32B chunk per lane.
  const unsigned short* a0p = wbf + (size_t)(wv * 32 + row) * CK + half * 16;
  const unsigned short* a1p = a0p + (size_t)16 * CK;
  // B in the same contiguous k order as A: lane holds col=lane%16, K = k0 + half*16 + j
  // => one contiguous 32B LDS chunk per lane.
  const unsigned short* bls = &colLDS[row * PADK + half * 16];

  for (int k0 = 0; k0 < CK; k0 += KSTEP) {
    v8u bu = *(const v8u*)(bls + k0);
    v16bf bF = __builtin_bit_cast(v16bf, bu);

    v4u lo0 = *(const v4u*)(a0p + k0);
    v4u hi0 = *(const v4u*)(a0p + k0 + 8);
    v4u lo1 = *(const v4u*)(a1p + k0);
    v4u hi1 = *(const v4u*)(a1p + k0 + 8);
    __builtin_prefetch(a0p + k0 + KSTEP, 0, 3);
    __builtin_prefetch(a1p + k0 + KSTEP, 0, 3);
    v8u au0, au1;
#pragma unroll
    for (int i = 0; i < 4; ++i) {
      au0[i] = lo0[i]; au0[i + 4] = hi0[i];
      au1[i] = lo1[i]; au1[i + 4] = hi1[i];
    }
    v16bf a0 = __builtin_bit_cast(v16bf, au0);
    v16bf a1 = __builtin_bit_cast(v16bf, au1);

    acc0 = __builtin_amdgcn_wmma_f32_16x16x32_bf16(false, a0, false, bF,
                                                   (short)0, acc0, false, false);
    acc1 = __builtin_amdgcn_wmma_f32_16x16x32_bf16(false, a1, false, bF,
                                                   (short)0, acc1, false, false);
  }

  // ---- Epilogue: D layout -> lane holds col=lane%16, VGPR r -> M = r + 8*half ----
  const int pixg = pix0 + row;
  const size_t outn = ((size_t)n * P_) << 12;
#pragma unroll
  for (int r = 0; r < 8; ++r) {
    int M0 = wv * 32 + 8 * half + r;
    int M1 = M0 + 16;
    out[outn + ((size_t)M0 << 12) + pixg] = acc0[r] + bias[M0];
    out[outn + ((size_t)M1 << 12) + pixg] = acc1[r] + bias[M1];
  }
}

// ---------------------------------------------------------------------------
extern "C" void kernel_launch(void* const* d_in, const int* in_sizes, int n_in,
                              void* d_out, int out_size, void* d_ws, size_t ws_size,
                              hipStream_t stream) {
  const float* x        = (const float*)d_in[0];
  const float* offset_w = (const float*)d_in[1];
  const float* offset_b = (const float*)d_in[2];
  const float* mask_w   = (const float*)d_in[3];
  const float* mask_b   = (const float*)d_in[4];
  const float* weights  = (const float*)d_in[5];
  const float* bias     = (const float*)d_in[6];
  float* out = (float*)d_out;

  // Workspace layout: [bf16 weights | offset maps | mask maps | NHWC x] (~21 MB)
  unsigned short* wbf = (unsigned short*)d_ws;
  float* offbuf  = (float*)((char*)d_ws + (size_t)P_ * CK * sizeof(unsigned short));
  float* maskbuf = offbuf + (size_t)N_ * 18 * L_;
  float* xT      = maskbuf + (size_t)N_ * 9 * L_;

  wconvert_kernel<<<(P_ * CK) / 256, 256, 0, stream>>>(weights, wbf);
  xpose_kernel<<<dim3(L_ / 64, N_), 256, 0, stream>>>(x, xT);
  offmask_conv_kernel<<<(N_ * L_) / 256, 256, 0, stream>>>(
      x, offset_w, offset_b, mask_w, mask_b, offbuf, maskbuf);
  dim3 grid(L_ / PIX_TILE, N_, 1);
  dcn_fused_kernel<<<grid, 256, 0, stream>>>(xT, wbf, offbuf, maskbuf, bias, out);
}
